// ComplexAttention_76184129896794
// MI455X (gfx1250) — matrix operand order repaired and checked by reading, not compiled, not run on hardware
//
#include <hip/hip_runtime.h>
#include <math.h>

// ---------------- problem constants ----------------
#define BATCH 16
#define SEQ   500
#define DMODEL 1024
#define HEADS 16
#define DHEAD 64
#define LPAD  512          // padded key length (32 tiles of 16)
#define MROWS 8000         // BATCH*SEQ
#define SCALE 0.125f       // 1/sqrt(64)

// ---------------- WMMA types ----------------
typedef __attribute__((ext_vector_type(16))) __bf16          v16bf;
typedef __attribute__((ext_vector_type(8)))  float           v8f;
typedef __attribute__((ext_vector_type(16))) unsigned short  us16;
typedef __attribute__((ext_vector_type(8)))  unsigned int    u32x8;

struct U16Pair { uint4 lo, hi; };

__device__ __forceinline__ unsigned short f2bf(float f) {
  __bf16 h = (__bf16)f;                         // hardware cvt on gfx1250
  return __builtin_bit_cast(unsigned short, h);
}
__device__ __forceinline__ float bf2f(unsigned short h) {
  return (float)__builtin_bit_cast(__bf16, h);
}

__device__ __forceinline__ v8f wmma_bf16(us16 a, us16 b, v8f c) {
  return __builtin_amdgcn_wmma_f32_16x16x32_bf16(
      false, __builtin_bit_cast(v16bf, a),
      false, __builtin_bit_cast(v16bf, b),
      (short)0, c, false, false);
}

// A-fragment: elements 0..7 = p[0..7], elements 8..15 = p[16..23]
// (caller passes p = row + k0 + half*8 -> K = k0+half*8+j and k0+16+half*8+j)
__device__ __forceinline__ us16 ld_a_pair(const unsigned short* p) {
  U16Pair t;
  t.lo = *(const uint4*)(p);
  t.hi = *(const uint4*)(p + 16);
  return __builtin_bit_cast(us16, t);
}
// B-fragment (caller passes p = col_base + k0 + half*16): elements 0..7 = K k0+half*8+j,
// elements 8..15 = K k0+16+half*8+j (same order as ld_a_pair)
__device__ __forceinline__ us16 ld_b16(const unsigned short* p) {
  const int hb = (threadIdx.x >> 4) & 1;
  U16Pair t;
  t.lo = *(const uint4*)(p - hb * 8);
  t.hi = *(const uint4*)(p + 16 - hb * 8);
  return __builtin_bit_cast(us16, t);
}
__device__ __forceinline__ us16 neg_bf16(us16 x) {
  u32x8 u = __builtin_bit_cast(u32x8, x);
  u ^= 0x80008000u;
  return __builtin_bit_cast(us16, u);
}
__device__ __forceinline__ v8f v8f_zero() {
  v8f z = {0.f,0.f,0.f,0.f,0.f,0.f,0.f,0.f};
  return z;
}

// =====================================================================
// Kernel 1: transpose + bf16-convert the 4 weight matrices: Wt[n][k]=W[k][n]
// =====================================================================
__global__ __launch_bounds__(256) void prep_weights(
    const float* __restrict__ Wk, const float* __restrict__ Wv,
    const float* __restrict__ Wq, const float* __restrict__ Wf,
    unsigned short* __restrict__ Tk, unsigned short* __restrict__ Tv,
    unsigned short* __restrict__ Tq, unsigned short* __restrict__ Tf) {
  size_t i = (size_t)blockIdx.x * 256 + threadIdx.x;   // 0 .. 4M-1
  int w = (int)(i >> 20);
  size_t r = i & 0xFFFFFu;                              // element in 1024x1024
  int k = (int)(r >> 10);
  int n = (int)(r & 1023);
  const float* W = (w == 0) ? Wk : (w == 1) ? Wv : (w == 2) ? Wq : Wf;
  unsigned short* T = (w == 0) ? Tk : (w == 1) ? Tv : (w == 2) ? Tq : Tf;
  T[(size_t)n * 1024 + k] = f2bf(W[r]);
}

// =====================================================================
// Kernel 1b: zero the key-pad region [500,512) of the transposed V buffers
// so padded B-fragment elements are exactly 0 (P=0 * 0 = 0, never NaN).
// =====================================================================
__global__ __launch_bounds__(256) void zero_vt_pad(
    unsigned short* __restrict__ vTr, unsigned short* __restrict__ vTi) {
  const int pad = LPAD - SEQ;                  // 12
  const int total = BATCH * HEADS * DHEAD * pad;
  int i = blockIdx.x * 256 + threadIdx.x;
  if (i < total) {
    int key = SEQ + (i % pad);
    int bhd = i / pad;
    size_t o = (size_t)bhd * LPAD + key;
    vTr[o] = 0;
    vTi[o] = 0;
  }
}

// =====================================================================
// Kernel 2: projection GEMM  Y = (X (* sin(phase) if USE_SIN)) @ W + b
// STORE_T=false: Y bf16 [8000,1024] row-major (K,Q projections, ctx inputs)
// STORE_T=true : Y bf16 [B][H][DHEAD][LPAD] head-transposed (V projections)
// Workgroup: 8 waves as 2(M)x4(N) -> 64x256 tile. Grid (125,4).
// =====================================================================
template <bool USE_SIN, bool STORE_T>
__global__ __launch_bounds__(256) void proj_gemm(
    const float* __restrict__ X, const unsigned short* __restrict__ Wt,
    const float* __restrict__ bias, const float* __restrict__ phase,
    unsigned short* __restrict__ Y) {
  const int tid = threadIdx.x;
  const int wave = tid >> 5, lane = tid & 31;
  const int half = lane >> 4, l16 = lane & 15;
  const int wm = wave & 1, wn = wave >> 1;
  const int M0 = blockIdx.x * 64 + wm * 32;
  const int N0 = blockIdx.y * 256 + wn * 64;

  v8f acc[2][4];
#pragma unroll
  for (int m = 0; m < 2; ++m)
#pragma unroll
    for (int n = 0; n < 4; ++n) acc[m][n] = v8f_zero();

  for (int k0 = 0; k0 < DMODEL; k0 += 32) {
    // prefetch next k-step (gfx1250 global_prefetch_b8)
    if (k0 + 32 < DMODEL) {
      __builtin_prefetch((const void*)(X + (size_t)(M0 + l16) * DMODEL + k0 + 32), 0, 1);
      __builtin_prefetch((const void*)(Wt + (size_t)(N0 + l16) * DMODEL + k0 + 32), 0, 1);
    }
    us16 afr[2];
#pragma unroll
    for (int mt = 0; mt < 2; ++mt) {
      int row = M0 + mt * 16 + l16;
      const float* rp = X + (size_t)row * DMODEL;
      const int b0 = k0 + half * 8;
      float4 x0 = *(const float4*)(rp + b0);
      float4 x1 = *(const float4*)(rp + b0 + 4);
      float4 x2 = *(const float4*)(rp + b0 + 16);
      float4 x3 = *(const float4*)(rp + b0 + 20);
      if (USE_SIN) {
        const float* sp = phase + (size_t)(row % SEQ) * DMODEL;
        float4 s0 = *(const float4*)(sp + b0);
        float4 s1 = *(const float4*)(sp + b0 + 4);
        float4 s2 = *(const float4*)(sp + b0 + 16);
        float4 s3 = *(const float4*)(sp + b0 + 20);
        x0.x *= __sinf(s0.x); x0.y *= __sinf(s0.y); x0.z *= __sinf(s0.z); x0.w *= __sinf(s0.w);
        x1.x *= __sinf(s1.x); x1.y *= __sinf(s1.y); x1.z *= __sinf(s1.z); x1.w *= __sinf(s1.w);
        x2.x *= __sinf(s2.x); x2.y *= __sinf(s2.y); x2.z *= __sinf(s2.z); x2.w *= __sinf(s2.w);
        x3.x *= __sinf(s3.x); x3.y *= __sinf(s3.y); x3.z *= __sinf(s3.z); x3.w *= __sinf(s3.w);
      }
      us16 r;
      r[0]  = f2bf(x0.x); r[1]  = f2bf(x0.y); r[2]  = f2bf(x0.z); r[3]  = f2bf(x0.w);
      r[4]  = f2bf(x1.x); r[5]  = f2bf(x1.y); r[6]  = f2bf(x1.z); r[7]  = f2bf(x1.w);
      r[8]  = f2bf(x2.x); r[9]  = f2bf(x2.y); r[10] = f2bf(x2.z); r[11] = f2bf(x2.w);
      r[12] = f2bf(x3.x); r[13] = f2bf(x3.y); r[14] = f2bf(x3.z); r[15] = f2bf(x3.w);
      afr[mt] = r;
    }
    us16 bfr[4];
#pragma unroll
    for (int nt = 0; nt < 4; ++nt) {
      int col = N0 + nt * 16 + l16;
      bfr[nt] = ld_b16(Wt + (size_t)col * DMODEL + k0 + half * 16);
    }
#pragma unroll
    for (int mt = 0; mt < 2; ++mt)
#pragma unroll
      for (int nt = 0; nt < 4; ++nt)
        acc[mt][nt] = wmma_bf16(afr[mt], bfr[nt], acc[mt][nt]);
  }

  // store bf16 with bias (C layout: VGPR i -> row i + 8*half, col = l16)
#pragma unroll
  for (int mt = 0; mt < 2; ++mt)
#pragma unroll
    for (int nt = 0; nt < 4; ++nt) {
      int col = N0 + nt * 16 + l16;
      float bv = bias[col];
#pragma unroll
      for (int i = 0; i < 8; ++i) {
        int row = M0 + mt * 16 + i + 8 * half;
        unsigned short v = f2bf(acc[mt][nt][i] + bv);
        if (STORE_T) {
          int bb = row / SEQ, l = row - bb * SEQ;
          int hh = col >> 6, d = col & 63;
          Y[((size_t)((bb * HEADS + hh) * DHEAD) + d) * LPAD + l] = v;
        } else {
          Y[(size_t)row * DMODEL + col] = v;
        }
      }
    }
}

// =====================================================================
// Kernel 3: dual-softmax complex attention, one WG per (qtile16, h, b).
//   s_real = qr.krT + qi.kiT ; s_imag = qi.krT + qr.(-ki)T (scaled)
//   softmax both (unnormalized exp in LDS), ctx = P@V via WMMA (V comes from
//   the head-transposed bf16 buffers -> contiguous B fragments), normalize on
//   write. Head 0 also emits |attn| to the d_out tail.
// =====================================================================
__global__ __launch_bounds__(256) void attn_kernel(
    const unsigned short* __restrict__ kr, const unsigned short* __restrict__ ki,
    const unsigned short* __restrict__ vTr, const unsigned short* __restrict__ vTi,
    const unsigned short* __restrict__ qr, const unsigned short* __restrict__ qi,
    unsigned short* __restrict__ ctx_r, unsigned short* __restrict__ ctx_i,
    float* __restrict__ attn_out) {
  __shared__ unsigned short Pr[16][LPAD];
  __shared__ unsigned short Pi[16][LPAD];
  __shared__ float red[2][8][16];
  __shared__ float rowmax[2][16];
  __shared__ float rowsum[2][16];
  __shared__ float ctxS[2][16][DHEAD];

  const int tid = threadIdx.x;
  const int wave = tid >> 5, lane = tid & 31;
  const int half = lane >> 4, l16 = lane & 15;
  const int qt = blockIdx.x, h = blockIdx.y, b = blockIdx.z;
  const int qbase = qt * 16;

  // zero ctx accumulator
  for (int i = tid; i < 2 * 16 * DHEAD; i += 256) ((float*)ctxS)[i] = 0.f;

  int qrow = qbase + l16; if (qrow > SEQ - 1) qrow = SEQ - 1;
  const unsigned short* qr_row = qr + ((size_t)(b * SEQ + qrow)) * DMODEL + h * DHEAD;
  const unsigned short* qi_row = qi + ((size_t)(b * SEQ + qrow)) * DMODEL + h * DHEAD;

  // ---- phase 1: scores via WMMA (wave owns key tiles wave*4 .. wave*4+3) ----
  v8f accR[4], accI[4];
#pragma unroll
  for (int j = 0; j < 4; ++j) { accR[j] = v8f_zero(); accI[j] = v8f_zero(); }

#pragma unroll
  for (int ks = 0; ks < 2; ++ks) {
    const int k0 = ks * 32;
    us16 aqr = ld_a_pair(qr_row + k0 + half * 8);
    us16 aqi = ld_a_pair(qi_row + k0 + half * 8);
#pragma unroll
    for (int j = 0; j < 4; ++j) {
      int key = (wave * 4 + j) * 16 + l16;
      int kc = key > SEQ - 1 ? SEQ - 1 : key;
      const unsigned short* krp = kr + ((size_t)(b * SEQ + kc)) * DMODEL + h * DHEAD;
      const unsigned short* kip = ki + ((size_t)(b * SEQ + kc)) * DMODEL + h * DHEAD;
      us16 bkr = ld_b16(krp + k0 + half * 16);
      us16 bki = ld_b16(kip + k0 + half * 16);
      us16 nki = neg_bf16(bki);
      accR[j] = wmma_bf16(aqr, bkr, accR[j]);
      accR[j] = wmma_bf16(aqi, bki, accR[j]);
      accI[j] = wmma_bf16(aqi, bkr, accI[j]);
      accI[j] = wmma_bf16(aqr, nki, accI[j]);
    }
  }

  // ---- row max (shuffle within lane-half, then cross-wave via LDS) ----
  float pm[2][8];
#pragma unroll
  for (int i = 0; i < 8; ++i) { pm[0][i] = -3.0e38f; pm[1][i] = -3.0e38f; }
#pragma unroll
  for (int j = 0; j < 4; ++j) {
    bool valid = (wave * 64 + j * 16 + l16) < SEQ;
    if (valid) {
#pragma unroll
      for (int i = 0; i < 8; ++i) {
        pm[0][i] = fmaxf(pm[0][i], accR[j][i]);
        pm[1][i] = fmaxf(pm[1][i], accI[j][i]);
      }
    }
  }
#pragma unroll
  for (int m = 1; m < 16; m <<= 1)
#pragma unroll
    for (int i = 0; i < 8; ++i) {
      pm[0][i] = fmaxf(pm[0][i], __shfl_xor(pm[0][i], m, 32));
      pm[1][i] = fmaxf(pm[1][i], __shfl_xor(pm[1][i], m, 32));
    }
  if (l16 == 0) {
#pragma unroll
    for (int i = 0; i < 8; ++i) {
      red[0][wave][half * 8 + i] = pm[0][i];
      red[1][wave][half * 8 + i] = pm[1][i];
    }
  }
  __syncthreads();
  if (tid < 32) {
    int c = tid >> 4, r = tid & 15;
    float m = -3.0e38f;
#pragma unroll
    for (int w = 0; w < 8; ++w) m = fmaxf(m, red[c][w][r]);
    rowmax[c][r] = m;
  }
  __syncthreads();

  // ---- exp (scaled), write P to LDS (bf16), partial row sums ----
  float psum[2][8];
#pragma unroll
  for (int i = 0; i < 8; ++i) { psum[0][i] = 0.f; psum[1][i] = 0.f; }
#pragma unroll
  for (int j = 0; j < 4; ++j) {
    int keyIdx = wave * 64 + j * 16 + l16;
    bool valid = keyIdx < SEQ;
#pragma unroll
    for (int i = 0; i < 8; ++i) {
      int row = i + 8 * half;
      float er = valid ? __expf((accR[j][i] - rowmax[0][row]) * SCALE) : 0.f;
      float ei = valid ? __expf((accI[j][i] - rowmax[1][row]) * SCALE) : 0.f;
      Pr[row][keyIdx] = f2bf(er);
      Pi[row][keyIdx] = f2bf(ei);
      psum[0][i] += er;
      psum[1][i] += ei;
    }
  }
#pragma unroll
  for (int m = 1; m < 16; m <<= 1)
#pragma unroll
    for (int i = 0; i < 8; ++i) {
      psum[0][i] += __shfl_xor(psum[0][i], m, 32);
      psum[1][i] += __shfl_xor(psum[1][i], m, 32);
    }
  if (l16 == 0) {
#pragma unroll
    for (int i = 0; i < 8; ++i) {
      red[0][wave][half * 8 + i] = psum[0][i];
      red[1][wave][half * 8 + i] = psum[1][i];
    }
  }
  __syncthreads();
  if (tid < 32) {
    int c = tid >> 4, r = tid & 15;
    float s = 0.f;
#pragma unroll
    for (int w = 0; w < 8; ++w) s += red[c][w][r];
    rowsum[c][r] = (s < 1e-30f) ? 1e-30f : s;
  }
  __syncthreads();

  // ---- phase 2: ctx = P @ V (wave owns key-ksteps wave*2, wave*2+1) ----
  v8f cR[4], cI[4];
#pragma unroll
  for (int n = 0; n < 4; ++n) { cR[n] = v8f_zero(); cI[n] = v8f_zero(); }
  const unsigned short* vtr_base = vTr + ((size_t)((b * HEADS + h) * DHEAD)) * LPAD;
  const unsigned short* vti_base = vTi + ((size_t)((b * HEADS + h) * DHEAD)) * LPAD;

#pragma unroll
  for (int kk = 0; kk < 2; ++kk) {
    const int kb = (wave * 2 + kk) * 32;
    us16 apr = ld_a_pair(&Pr[l16][kb + half * 8]);
    us16 api = ld_a_pair(&Pi[l16][kb + half * 8]);
#pragma unroll
    for (int nt = 0; nt < 4; ++nt) {
      int col = nt * 16 + l16;   // d index
      us16 bvr = ld_b16(vtr_base + (size_t)col * LPAD + kb + half * 16);
      us16 bvi = ld_b16(vti_base + (size_t)col * LPAD + kb + half * 16);
      cR[nt] = wmma_bf16(apr, bvr, cR[nt]);
      cI[nt] = wmma_bf16(api, bvi, cI[nt]);
    }
  }
#pragma unroll
  for (int nt = 0; nt < 4; ++nt)
#pragma unroll
    for (int i = 0; i < 8; ++i) {
      int row = i + 8 * half, col = nt * 16 + l16;
      atomicAdd(&ctxS[0][row][col], cR[nt][i]);
      atomicAdd(&ctxS[1][row][col], cI[nt][i]);
    }
  __syncthreads();

  // ---- phase 3: normalized writes ----
  for (int idx = tid; idx < 16 * DHEAD; idx += 256) {
    int r = idx >> 6, d = idx & 63;
    int q = qbase + r;
    if (q < SEQ) {
      size_t o = ((size_t)(b * SEQ + q)) * DMODEL + h * DHEAD + d;
      ctx_r[o] = f2bf(ctxS[0][r][d] / rowsum[0][r]);
      ctx_i[o] = f2bf(ctxS[1][r][d] / rowsum[1][r]);
    }
  }
  if (h == 0) {
    for (int idx = tid; idx < 16 * SEQ; idx += 256) {
      int r = idx / SEQ, k = idx % SEQ;
      int q = qbase + r;
      if (q < SEQ) {
        float er = bf2f(Pr[r][k]) / rowsum[0][r];
        float ei = bf2f(Pi[r][k]) / rowsum[1][r];
        attn_out[((size_t)(b * SEQ + q)) * SEQ + k] = sqrtf(er * er + ei * ei);
      }
    }
  }
}

// =====================================================================
// Kernel 4: out = sqrt((ctx_r@Wf+bf)^2 + (ctx_i@Wf+bf)^2). Grid (125,4).
// =====================================================================
__global__ __launch_bounds__(256) void out_gemm(
    const unsigned short* __restrict__ Ar, const unsigned short* __restrict__ Ai,
    const unsigned short* __restrict__ Wt, const float* __restrict__ bias,
    float* __restrict__ out) {
  const int tid = threadIdx.x;
  const int wave = tid >> 5, lane = tid & 31;
  const int half = lane >> 4, l16 = lane & 15;
  const int wm = wave & 1, wn = wave >> 1;
  const int M0 = blockIdx.x * 64 + wm * 32;
  const int N0 = blockIdx.y * 256 + wn * 64;

  v8f accR[2][4], accI[2][4];
#pragma unroll
  for (int m = 0; m < 2; ++m)
#pragma unroll
    for (int n = 0; n < 4; ++n) { accR[m][n] = v8f_zero(); accI[m][n] = v8f_zero(); }

  for (int k0 = 0; k0 < DMODEL; k0 += 32) {
    if (k0 + 32 < DMODEL) {
      __builtin_prefetch((const void*)(Ar + (size_t)(M0 + l16) * DMODEL + k0 + 32), 0, 1);
      __builtin_prefetch((const void*)(Wt + (size_t)(N0 + l16) * DMODEL + k0 + 32), 0, 1);
    }
    us16 ar[2], ai[2];
#pragma unroll
    for (int mt = 0; mt < 2; ++mt) {
      int row = M0 + mt * 16 + l16;
      ar[mt] = ld_a_pair(Ar + (size_t)row * DMODEL + k0 + half * 8);
      ai[mt] = ld_a_pair(Ai + (size_t)row * DMODEL + k0 + half * 8);
    }
    us16 bfr[4];
#pragma unroll
    for (int nt = 0; nt < 4; ++nt) {
      int col = N0 + nt * 16 + l16;
      bfr[nt] = ld_b16(Wt + (size_t)col * DMODEL + k0 + half * 16);
    }
#pragma unroll
    for (int mt = 0; mt < 2; ++mt)
#pragma unroll
      for (int nt = 0; nt < 4; ++nt) {
        accR[mt][nt] = wmma_bf16(ar[mt], bfr[nt], accR[mt][nt]);
        accI[mt][nt] = wmma_bf16(ai[mt], bfr[nt], accI[mt][nt]);
      }
  }
#pragma unroll
  for (int mt = 0; mt < 2; ++mt)
#pragma unroll
    for (int nt = 0; nt < 4; ++nt) {
      int col = N0 + nt * 16 + l16;
      float bv = bias[col];
#pragma unroll
      for (int i = 0; i < 8; ++i) {
        int row = M0 + mt * 16 + i + 8 * half;
        float vrr = accR[mt][nt][i] + bv;
        float vii = accI[mt][nt][i] + bv;
        out[(size_t)row * DMODEL + col] = sqrtf(vrr * vrr + vii * vii);
      }
    }
}

// =====================================================================
// kernel_launch
// =====================================================================
extern "C" void kernel_launch(void* const* d_in, const int* in_sizes, int n_in,
                              void* d_out, int out_size, void* d_ws, size_t ws_size,
                              hipStream_t stream) {
  (void)in_sizes; (void)n_in; (void)out_size; (void)ws_size;
  const float* key   = (const float*)d_in[0];
  const float* value = (const float*)d_in[1];
  const float* query = (const float*)d_in[2];
  const float* phase = (const float*)d_in[3];
  const float* Wk = (const float*)d_in[4];
  const float* bk = (const float*)d_in[5];
  const float* Wv = (const float*)d_in[6];
  const float* bv = (const float*)d_in[7];
  const float* Wq = (const float*)d_in[8];
  const float* bq = (const float*)d_in[9];
  const float* Wf = (const float*)d_in[10];
  const float* bf_ = (const float*)d_in[11];

  // workspace layout (bytes):
  //   4 transposed bf16 weights (4 * 2 MiB)
  //   4 bf16 projections kr,ki,qr,qi (4 * 16,384,000)
  //   2 head-transposed bf16 V buffers (2 * 16,777,216)
  //   2 bf16 context buffers (2 * 16,384,000)
  //   total ~140 MB
  char* ws = (char*)d_ws;
  const size_t WSZ = (size_t)DMODEL * DMODEL * 2;                 // 2 MiB
  const size_t PSZ = (size_t)MROWS * DMODEL * 2;                  // 16,384,000 B
  const size_t VSZ = (size_t)BATCH * HEADS * DHEAD * LPAD * 2;    // 16,777,216 B
  unsigned short* tWk = (unsigned short*)(ws + 0 * WSZ);
  unsigned short* tWv = (unsigned short*)(ws + 1 * WSZ);
  unsigned short* tWq = (unsigned short*)(ws + 2 * WSZ);
  unsigned short* tWf = (unsigned short*)(ws + 3 * WSZ);
  char* p = ws + 4 * WSZ;
  unsigned short* pkr = (unsigned short*)(p);
  unsigned short* pki = (unsigned short*)(p + 1 * PSZ);
  unsigned short* pqr = (unsigned short*)(p + 2 * PSZ);
  unsigned short* pqi = (unsigned short*)(p + 3 * PSZ);
  unsigned short* vTr = (unsigned short*)(p + 4 * PSZ);
  unsigned short* vTi = (unsigned short*)(p + 4 * PSZ + VSZ);
  unsigned short* cxr = (unsigned short*)(p + 4 * PSZ + 2 * VSZ);
  unsigned short* cxi = (unsigned short*)(p + 4 * PSZ + 2 * VSZ + PSZ);

  float* out  = (float*)d_out;                       // [16,500,1024]
  float* attn = out + (size_t)MROWS * DMODEL;        // [16,500,500]

  prep_weights<<<(4u * 1024 * 1024) / 256, 256, 0, stream>>>(
      Wk, Wv, Wq, Wf, tWk, tWv, tWq, tWf);
  zero_vt_pad<<<(BATCH * HEADS * DHEAD * (LPAD - SEQ) + 255) / 256, 256, 0, stream>>>(
      vTr, vTi);

  dim3 gemmGrid(MROWS / 64, DMODEL / 256);
  proj_gemm<false, false><<<gemmGrid, 256, 0, stream>>>(key,   tWk, bk, phase, pkr);
  proj_gemm<true,  false><<<gemmGrid, 256, 0, stream>>>(key,   tWk, bk, phase, pki);
  proj_gemm<false, true ><<<gemmGrid, 256, 0, stream>>>(value, tWv, bv, phase, vTr);
  proj_gemm<true,  true ><<<gemmGrid, 256, 0, stream>>>(value, tWv, bv, phase, vTi);
  proj_gemm<false, false><<<gemmGrid, 256, 0, stream>>>(query, tWq, bq, phase, pqr);
  proj_gemm<true,  false><<<gemmGrid, 256, 0, stream>>>(query, tWq, bq, phase, pqi);

  attn_kernel<<<dim3(LPAD / 16, HEADS, BATCH), 256, 0, stream>>>(
      pkr, pki, vTr, vTi, pqr, pqi, cxr, cxi, attn);

  out_gemm<<<gemmGrid, 256, 0, stream>>>(cxr, cxi, tWf, bf_, out);
}
